// MobileNetV1Qua_39006892982867
// MI455X (gfx1250) — 8-bit matrix operand order repaired and checked by reading, compiled, not run on hardware
//
#include <hip/hip_runtime.h>
#include <stdint.h>

typedef __attribute__((ext_vector_type(8))) int v8i;

#define EPSF 1e-5f

static const int LK[15]     = {0, 1, 2, 1, 2, 1, 2, 1, 2, 1, 2, 1, 2, 1, 2};
static const int LCIN[15]   = {3, 32, 32, 64, 64, 128,128,128,128,256,256,512,512,512,512};
static const int LCOUT[15]  = {32,32, 64, 64, 128,128,128,128,256,256,512,512,512,512,1024};
static const int LSTRIDE[15]= {2, 1, 1, 2, 1, 1, 1, 2, 1, 2, 1, 1, 1, 2, 1};

__global__ void init_maxabs_kernel(unsigned* slots) {
    if (threadIdx.x < 32) slots[threadIdx.x] = 0u;
}

__global__ void reduce_maxabs_kernel(const float* __restrict__ w, long n, unsigned* slot) {
    long i = blockIdx.x * (long)blockDim.x + threadIdx.x;
    long stride = (long)gridDim.x * blockDim.x;
    unsigned local = 0u;
    for (; i < n; i += stride) {
        unsigned v = __float_as_uint(fabsf(tanhf(w[i])));
        local = (v > local) ? v : local;
    }
    #pragma unroll
    for (int off = 16; off > 0; off >>= 1) {
        unsigned o = __shfl_xor(local, off, 32);
        local = (o > local) ? o : local;
    }
    if ((threadIdx.x & 31) == 0) atomicMax(slot, local);
}

__device__ __forceinline__ int8_t quant_w4(float w, float maxabs) {
    float t = tanhf(w);
    float v01 = t / (2.0f * maxabs) + 0.5f;
    int q = (int)rintf(v01 * 15.0f);
    return (int8_t)(2 * q - 15);
}

__global__ void quant_small_kernel(const float* __restrict__ w, int n,
                                   const unsigned* slot, int8_t* out) {
    int i = blockIdx.x * blockDim.x + threadIdx.x;
    if (i >= n) return;
    out[i] = quant_w4(w[i], __uint_as_float(*slot));
}

__global__ void pack_b_kernel(const float* __restrict__ w, const unsigned* slot,
                              int8_t* __restrict__ packed,
                              int Cin, int Cout, int Ktiles, int Ntiles) {
    int total = Ntiles * Ktiles * 1024;
    int idx = blockIdx.x * blockDim.x + threadIdx.x;
    if (idx >= total) return;
    int byteInTile = idx & 1023;
    int tile = idx >> 10;
    int kt = tile % Ktiles;
    int nt = tile / Ktiles;
    int lane = byteInTile >> 5;
    int r = byteInTile & 31;
    int v = r >> 2, bb = r & 3;
    int n = nt * 16 + (lane & 15);
    int k = kt * 64 + ((v >> 1) << 4) + ((lane >> 4) << 3) + ((v & 1) << 2) + bb;
    int8_t val = 0;
    if (n < Cout && k < Cin)
        val = quant_w4(w[(size_t)n * Cin + k], __uint_as_float(*slot));
    packed[idx] = val;
}

__global__ void first_conv_kernel(const float* __restrict__ x, const int8_t* __restrict__ w8,
                                  const float* __restrict__ g, const float* __restrict__ b_,
                                  const float* __restrict__ mu, const float* __restrict__ var,
                                  uint8_t* __restrict__ out) {
    const int H = 112;
    int idx = blockIdx.x * blockDim.x + threadIdx.x;
    int total = 32 * H * H * 32;
    if (idx >= total) return;
    int co = idx & 31;
    int t = idx >> 5;
    int wo = t % H; t /= H;
    int ho = t % H; int n = t / H;
    float sum = 0.0f;
    #pragma unroll
    for (int ci = 0; ci < 3; ++ci)
        #pragma unroll
        for (int kh = 0; kh < 3; ++kh) {
            int ih = ho * 2 + kh - 1;
            if ((unsigned)ih >= 224u) continue;
            #pragma unroll
            for (int kw = 0; kw < 3; ++kw) {
                int iw = wo * 2 + kw - 1;
                if ((unsigned)iw >= 224u) continue;
                float xv = x[(((size_t)n * 3 + ci) * 224 + ih) * 224 + iw];
                sum += xv * (float)w8[((co * 3 + ci) * 3 + kh) * 3 + kw];
            }
        }
    float scale = g[co] * rsqrtf(var[co] + EPSF);
    float bias = b_[co] - mu[co] * scale;
    float y = sum * (1.0f / 15.0f) * scale + bias;
    y = fminf(fmaxf(y, 0.0f), 1.0f);
    out[idx] = (uint8_t)(int)rintf(y * 15.0f);
}

__global__ void dw_kernel(const uint8_t* __restrict__ in, const int8_t* __restrict__ w9,
                          const float* __restrict__ g, const float* __restrict__ b_,
                          const float* __restrict__ mu, const float* __restrict__ var,
                          uint8_t* __restrict__ out, int C, int Cst,
                          int Hin, int Hout, int stride) {
    long idx = blockIdx.x * (long)blockDim.x + threadIdx.x;
    long total = (long)32 * Hout * Hout * Cst;
    if (idx >= total) return;
    int c = (int)(idx % Cst); long t = idx / Cst;
    if (c >= C) { out[idx] = 0; return; }
    int wo = (int)(t % Hout); t /= Hout;
    int ho = (int)(t % Hout); int n = (int)(t / Hout);
    int sum = 0;
    #pragma unroll
    for (int kh = 0; kh < 3; ++kh) {
        int ih = ho * stride + kh - 1;
        if ((unsigned)ih >= (unsigned)Hin) continue;
        #pragma unroll
        for (int kw = 0; kw < 3; ++kw) {
            int iw = wo * stride + kw - 1;
            if ((unsigned)iw >= (unsigned)Hin) continue;
            sum += (int)in[(((size_t)n * Hin + ih) * Hin + iw) * C + c]
                 * (int)w9[c * 9 + kh * 3 + kw];
        }
    }
    float scale = g[c] * rsqrtf(var[c] + EPSF);
    float bias = b_[c] - mu[c] * scale;
    float y = (float)sum * (1.0f / 225.0f) * scale + bias;
    y = fminf(fmaxf(y, 0.0f), 1.0f);
    out[idx] = (uint8_t)(int)rintf(y * 15.0f);
}

__device__ __forceinline__ void wave_gemm_iu8_x4(const uint8_t* __restrict__ A, int ldA,
                                                 const int8_t* __restrict__ Bp,
                                                 int Ktiles, int nt0, int rowBase,
                                                 v8i acc[4]) {
    int lane = threadIdx.x & 31;
    int lh = lane >> 4;
    const uint8_t* arow = A + (size_t)(rowBase + (lane & 15)) * ldA + (lh << 3);
    const int8_t* bbase = Bp + ((size_t)nt0 * Ktiles << 10) + (lane << 5);
    for (int kt = 0; kt < Ktiles; ++kt) {
        const uint8_t* ap = arow + (kt << 6);
        __builtin_prefetch(ap + 64, 0, 1);
        uint2 a01 = *(const uint2*)(ap);
        uint2 a23 = *(const uint2*)(ap + 16);
        uint2 a45 = *(const uint2*)(ap + 32);
        uint2 a67 = *(const uint2*)(ap + 48);
        v8i a;
        a[0] = (int)a01.x; a[1] = (int)a01.y;
        a[2] = (int)a23.x; a[3] = (int)a23.y;
        a[4] = (int)a45.x; a[5] = (int)a45.y;
        a[6] = (int)a67.x; a[7] = (int)a67.y;
        #pragma unroll
        for (int j = 0; j < 4; ++j) {
            const int4* bp = (const int4*)(bbase + ((size_t)(j * Ktiles + kt) << 10));
            int4 b0 = bp[0];
            int4 b1 = bp[1];
            v8i b;
            b[0] = b0.x; b[1] = b0.y; b[2] = b0.z; b[3] = b0.w;
            b[4] = b1.x; b[5] = b1.y; b[6] = b1.z; b[7] = b1.w;
            acc[j] = __builtin_amdgcn_wmma_i32_16x16x64_iu8(false, a, true, b,
                                                            acc[j], false, false);
        }
    }
}

__global__ void pw_kernel(const uint8_t* __restrict__ in, const int8_t* __restrict__ packed,
                          const float* __restrict__ g, const float* __restrict__ b_,
                          const float* __restrict__ mu, const float* __restrict__ var,
                          uint8_t* __restrict__ out, int M, int Cin, int Cout) {
    int tilesN4 = Cout >> 6;
    int tilesM = M >> 4;
    int waveId = blockIdx.x * (blockDim.x >> 5) + (threadIdx.x >> 5);
    if (waveId >= tilesM * tilesN4) return;
    int nt4 = waveId % tilesN4;
    int mt = waveId / tilesN4;
    int Ktiles = (Cin + 63) >> 6;
    v8i acc[4] = {{0,0,0,0,0,0,0,0}, {0,0,0,0,0,0,0,0},
                  {0,0,0,0,0,0,0,0}, {0,0,0,0,0,0,0,0}};
    wave_gemm_iu8_x4(in, Ktiles << 6, packed, Ktiles, nt4 << 2, mt << 4, acc);
    int lane = threadIdx.x & 31;
    #pragma unroll
    for (int j = 0; j < 4; ++j) {
        int n = ((nt4 << 2) + j) * 16 + (lane & 15);
        float scale = g[n] * rsqrtf(var[n] + EPSF);
        float bias = b_[n] - mu[n] * scale;
        scale *= (1.0f / 225.0f);
        #pragma unroll
        for (int d = 0; d < 8; ++d) {
            int m = (mt << 4) + d + ((lane >> 4) << 3);
            float y = (float)acc[j][d] * scale + bias;
            y = fminf(fmaxf(y, 0.0f), 1.0f);
            out[(size_t)m * Cout + n] = (uint8_t)(int)rintf(y * 15.0f);
        }
    }
}

__global__ void pool_kernel(const uint8_t* __restrict__ in, uint8_t* __restrict__ pooled) {
    int idx = blockIdx.x * blockDim.x + threadIdx.x;
    if (idx >= 32 * 1024) return;
    int c = idx & 1023;
    int n = idx >> 10;
    int sum = 0;
    #pragma unroll
    for (int h = 0; h < 7; ++h)
        #pragma unroll
        for (int w = 0; w < 7; ++w)
            sum += in[(((size_t)n * 7 + h) * 7 + w) * 1024 + c];
    float y = (float)sum * (1.0f / 480.0f);
    y = fminf(fmaxf(y, 0.0f), 1.0f);
    pooled[idx] = (uint8_t)(int)rintf(y * 31.0f);
}

__global__ void fc_kernel(const uint8_t* __restrict__ pooled, const int8_t* __restrict__ packed,
                          const float* __restrict__ fcb, float* __restrict__ out) {
    const int N = 1000;
    const int tilesN4 = 16, Ktiles = 16;
    int waveId = blockIdx.x * (blockDim.x >> 5) + (threadIdx.x >> 5);
    if (waveId >= 2 * tilesN4) return;
    int nt4 = waveId % tilesN4;
    int mt = waveId / tilesN4;
    v8i acc[4] = {{0,0,0,0,0,0,0,0}, {0,0,0,0,0,0,0,0},
                  {0,0,0,0,0,0,0,0}, {0,0,0,0,0,0,0,0}};
    wave_gemm_iu8_x4(pooled, 1024, packed, Ktiles, nt4 << 2, mt << 4, acc);
    int lane = threadIdx.x & 31;
    #pragma unroll
    for (int j = 0; j < 4; ++j) {
        int n = ((nt4 << 2) + j) * 16 + (lane & 15);
        #pragma unroll
        for (int d = 0; d < 8; ++d) {
            int m = (mt << 4) + d + ((lane >> 4) << 3);
            if (n < N)
                out[(size_t)m * N + n] = (float)acc[j][d] * (1.0f / (31.0f * 15.0f)) + fcb[n];
        }
    }
}

extern "C" void kernel_launch(void* const* d_in, const int* in_sizes, int n_in,
                              void* d_out, int out_size, void* d_ws, size_t ws_size,
                              hipStream_t stream) {
    (void)in_sizes; (void)n_in; (void)out_size; (void)ws_size;
    const float* x = (const float*)d_in[0];
    const float* convw[15];
    const float *ga[15], *be[15], *mu[15], *va[15];
    for (int i = 0; i < 15; ++i) {
        convw[i] = (const float*)d_in[1 + i];
        ga[i] = (const float*)d_in[16 + i];
        be[i] = (const float*)d_in[31 + i];
        mu[i] = (const float*)d_in[46 + i];
        va[i] = (const float*)d_in[61 + i];
    }
    const float* fcw = (const float*)d_in[76];
    const float* fcb = (const float*)d_in[77];

    const size_t ACT_BYTES = (size_t)32 * 112 * 112 * 64;
    uint8_t* ws = (uint8_t*)d_ws;
    uint8_t* actA = ws;
    uint8_t* actB = ws + ACT_BYTES;
    int8_t*  packedBase = (int8_t*)(ws + 2 * ACT_BYTES);
    uint8_t* smallBase = ws + 2 * ACT_BYTES + (4u << 20);
    uint8_t* pooled    = ws + 2 * ACT_BYTES + (4u << 20) + (64u << 10);
    unsigned* maxabs   = (unsigned*)(ws + 2 * ACT_BYTES + (4u << 20) + (128u << 10));

    init_maxabs_kernel<<<1, 32, 0, stream>>>(maxabs);

    for (int i = 0; i < 15; ++i) {
        long n;
        if (LK[i] == 0)      n = (long)LCOUT[i] * 3 * 9;
        else if (LK[i] == 1) n = (long)LCIN[i] * 9;
        else                 n = (long)LCOUT[i] * LCIN[i];
        int blocks = (int)((n + 255) / 256); if (blocks > 1024) blocks = 1024;
        reduce_maxabs_kernel<<<blocks, 256, 0, stream>>>(convw[i], n, maxabs + i);
    }
    reduce_maxabs_kernel<<<1024, 256, 0, stream>>>(fcw, 1000L * 1024, maxabs + 15);

    int8_t* pwPacked[15] = {};
    size_t packedOff = 0;
    int8_t* smallW[15] = {};
    size_t smallOff = 0;
    for (int i = 0; i < 15; ++i) {
        if (LK[i] == 2) {
            int Ntiles = LCOUT[i] >> 4;
            int Ktiles = (LCIN[i] + 63) >> 6;
            int total = Ntiles * Ktiles * 1024;
            pwPacked[i] = packedBase + packedOff;
            packedOff += (size_t)total;
            pack_b_kernel<<<(total + 255) / 256, 256, 0, stream>>>(
                convw[i], maxabs + i, pwPacked[i], LCIN[i], LCOUT[i], Ktiles, Ntiles);
        } else {
            int n = (LK[i] == 0) ? (32 * 3 * 9) : (LCIN[i] * 9);
            smallW[i] = (int8_t*)(smallBase + smallOff);
            smallOff += (size_t)((n + 255) & ~255);
            quant_small_kernel<<<(n + 255) / 256, 256, 0, stream>>>(
                convw[i], n, maxabs + i, smallW[i]);
        }
    }
    int8_t* fcPacked = packedBase + packedOff;
    {
        int Ntiles = 64, Ktiles = 16;
        int total = Ntiles * Ktiles * 1024;
        pack_b_kernel<<<(total + 255) / 256, 256, 0, stream>>>(
            fcw, maxabs + 15, fcPacked, 1024, 1000, Ktiles, Ntiles);
    }

    {
        int total = 32 * 112 * 112 * 32;
        first_conv_kernel<<<(total + 255) / 256, 256, 0, stream>>>(
            x, smallW[0], ga[0], be[0], mu[0], va[0], actA);
    }

    uint8_t* cur = actA;
    uint8_t* oth = actB;
    int H = 112;
    for (int i = 1; i < 15; ++i) {
        if (LK[i] == 1) {
            int C = LCIN[i];
            int Cst = (C < 64) ? 64 : C;
            int Hout = (LSTRIDE[i] == 2) ? (H >> 1) : H;
            long total = (long)32 * Hout * Hout * Cst;
            dw_kernel<<<(int)((total + 255) / 256), 256, 0, stream>>>(
                cur, smallW[i], ga[i], be[i], mu[i], va[i],
                oth, C, Cst, H, Hout, LSTRIDE[i]);
            H = Hout;
        } else {
            int M = 32 * H * H;
            int waves = (M >> 4) * (LCOUT[i] >> 6);
            int blocks = (waves + 7) / 8;
            pw_kernel<<<blocks, 256, 0, stream>>>(
                cur, pwPacked[i], ga[i], be[i], mu[i], va[i],
                oth, M, LCIN[i], LCOUT[i]);
        }
        uint8_t* t = cur; cur = oth; oth = t;
    }

    pool_kernel<<<(32 * 1024 + 255) / 256, 256, 0, stream>>>(cur, pooled);

    fc_kernel<<<(2 * 16 + 7) / 8, 256, 0, stream>>>(pooled, fcPacked, fcb, (float*)d_out);
}
